// KTGPTBlock_10986526343383
// MI455X (gfx1250) — compile-verified, matrix operand order checked by reading, not run on hardware
//
#include <hip/hip_runtime.h>
#include <math.h>

typedef __attribute__((ext_vector_type(16))) __bf16 v16bf;
typedef __attribute__((ext_vector_type(8)))  float  v8f;
typedef unsigned short u16;

#ifndef __has_builtin
#define __has_builtin(x) 0
#endif
#if defined(__AMDGCN__) && __has_builtin(__builtin_amdgcn_tensor_load_to_lds)
#define CDNA5_HAS_TDM 1
#else
#define CDNA5_HAS_TDM 0
#endif

__device__ __forceinline__ u16 f2bf(float f) {
  unsigned int u = __float_as_uint(f);
  unsigned int r = (u + 0x7FFFu + ((u >> 16) & 1u)) >> 16;
  return (u16)r;
}

#if CDNA5_HAS_TDM
typedef __attribute__((ext_vector_type(4))) unsigned int u32x4;
typedef __attribute__((ext_vector_type(8))) int i32x8;
typedef __attribute__((ext_vector_type(4))) int i32x4;

// Tensor Data Mover: 2D tile (2-byte elements) from global -> LDS.
// D# layout per CDNA5 ISA:
//   group0: count=1 | lds_addr | global_addr(57b) | type=2
//   group1: data_size=1(2B), tensor_dim0/1, tile_dim0/1, tensor_dim0_stride
__device__ __forceinline__ void tdm_load_2d(unsigned lds_addr, const void* gptr,
                                            unsigned tile_d0, unsigned tile_d1,
                                            unsigned tensor_d0, unsigned tensor_d1,
                                            unsigned stride0) {
  unsigned long long ga = (unsigned long long)gptr;
  u32x4 g0;
  g0[0] = 1u;                                             // count=1 (user D#)
  g0[1] = lds_addr;                                       // LDS byte address
  g0[2] = (unsigned)ga;                                   // global_addr[31:0]
  g0[3] = (unsigned)((ga >> 32) & 0x01FFFFFFu) | 0x80000000u; // addr[56:32] | type=2
  i32x8 g1;
  g1[0] = 0x00010000;                                     // data_size=1 (2 bytes)
  g1[1] = (int)((tensor_d0 & 0xFFFFu) << 16);             // tensor_dim0[15:0]
  g1[2] = (int)(((tensor_d0 >> 16) & 0xFFFFu) |
                ((tensor_d1 & 0xFFFFu) << 16));           // td0 hi | td1 lo
  g1[3] = (int)(((tensor_d1 >> 16) & 0xFFFFu) |
                ((tile_d0 & 0xFFFFu) << 16));             // td1 hi | tile_dim0
  g1[4] = (int)(tile_d1 & 0xFFFFu);                       // tile_dim1, tile_dim2=0
  g1[5] = (int)stride0;                                   // tensor_dim0_stride lo
  g1[6] = 0;
  g1[7] = 0;
  i32x4 z4; z4[0] = 0; z4[1] = 0; z4[2] = 0; z4[3] = 0;
#if __clang_major__ >= 23
  i32x8 z8; z8[0]=0; z8[1]=0; z8[2]=0; z8[3]=0; z8[4]=0; z8[5]=0; z8[6]=0; z8[7]=0;
  __builtin_amdgcn_tensor_load_to_lds(g0, g1, z4, z4, z8, 0);
#else
  __builtin_amdgcn_tensor_load_to_lds(g0, g1, z4, z4, 0);
#endif
}
#endif // CDNA5_HAS_TDM

// ---------------------------------------------------------------------------
// Generic bf16 WMMA GEMM:  C[M,N] = A[M,K] * B[K,N]  (+ optional residual,
// optional fused silu(gate)*out for the FFN up-projection).
// Block tile 128x128, KT=64, 8 waves; each wave 32x64 -> 16 wmma per K stage.
// A tile staged by TDM, double-buffered with issue-ahead (s_wait_tensorcnt 1).
// Requires: M%128==0, K%64==0, N%8==0.
// ---------------------------------------------------------------------------
__global__ __launch_bounds__(256)
void gemm_bf16_wmma(const u16* __restrict__ A, const u16* __restrict__ Bm,
                    int M, int N, int K,
                    const float* __restrict__ resid,
                    float* __restrict__ outF, u16* __restrict__ outB,
                    const float* __restrict__ gateMul)
{
  constexpr int KT = 64;
  __shared__ u16 As[2][128 * KT];    // ping-pong A tiles [row][k]
  __shared__ u16 Bs[128 * KT];       // transposed: [n][k]
  const int tid   = threadIdx.x;
  const int wid   = tid >> 5;
  const int lane  = tid & 31;
  const int bm    = blockIdx.y * 128;
  const int bn    = blockIdx.x * 128;
  const int waveM = (wid & 3) * 32;
  const int waveN = (wid >> 2) * 64;
  const int mrow  = lane & 15;
  const int khalf = (lane >> 4) * 16;

  v8f acc[2][4];
#pragma unroll
  for (int i = 0; i < 2; ++i)
#pragma unroll
    for (int j = 0; j < 4; ++j)
#pragma unroll
      for (int r = 0; r < 8; ++r) acc[i][j][r] = 0.0f;

#if CDNA5_HAS_TDM
  if (wid == 0)   // pre-issue first A tile; overlapped with B staging below
    tdm_load_2d((unsigned)(size_t)(void*)As[0], A + (size_t)bm * K,
                KT, 128, (unsigned)K, (unsigned)M, (unsigned)K);
#endif

  for (int k0 = 0; k0 < K; k0 += KT) {
    const int buf = (k0 / KT) & 1;
    // ---- stage A tile: 128 x 64 bf16 via TDM (issue-ahead, wait oldest) ----
#if CDNA5_HAS_TDM
    if (wid == 0) {
      if (k0 + KT < K) {
        tdm_load_2d((unsigned)(size_t)(void*)As[buf ^ 1],
                    A + (size_t)bm * K + (k0 + KT),
                    KT, 128, (unsigned)K, (unsigned)M, (unsigned)K);
        __builtin_amdgcn_s_wait_tensorcnt(1);   // tile k0 done, k0+KT in flight
      } else {
        __builtin_amdgcn_s_wait_tensorcnt(0);
      }
    }
#else
#pragma unroll
    for (int it = 0; it < 4; ++it) {
      int c   = tid + it * 256;
      int row = c >> 3;
      int col = (c & 7) * 8;
      *reinterpret_cast<uint4*>(&As[buf][row * KT + col]) =
          *reinterpret_cast<const uint4*>(A + (size_t)(bm + row) * K + (k0 + col));
    }
#endif
    // ---- stage B tile transposed: branch-free clamped loads ----
#pragma unroll
    for (int it = 0; it < 4; ++it) {
      int c  = tid + it * 256;
      int kk = c >> 4;
      int nn = (c & 15) * 8;
      int gn = bn + nn;
      int gns = (gn + 8 <= N) ? gn : (N - 8);   // N%8==0 -> only dead cols clamp
      u16 tmp[8];
      *reinterpret_cast<uint4*>(tmp) =
          *reinterpret_cast<const uint4*>(Bm + (size_t)(k0 + kk) * N + gns);
#pragma unroll
      for (int j = 0; j < 8; ++j) Bs[(nn + j) * KT + kk] = tmp[j];
    }
    if (k0 + KT < K)
      __builtin_prefetch(Bm + (size_t)(k0 + KT + (tid & 63)) * N + bn, 0, 1);
    __syncthreads();

    v16bf af[2][2], bfg[4][2];
#pragma unroll
    for (int i = 0; i < 2; ++i)
#pragma unroll
      for (int k2 = 0; k2 < 2; ++k2)
        af[i][k2] = *reinterpret_cast<const v16bf*>(
            &As[buf][(waveM + i * 16 + mrow) * KT + k2 * 32 + khalf]);
#pragma unroll
    for (int j = 0; j < 4; ++j)
#pragma unroll
      for (int k2 = 0; k2 < 2; ++k2)
        bfg[j][k2] = *reinterpret_cast<const v16bf*>(
            &Bs[(waveN + j * 16 + mrow) * KT + k2 * 32 + khalf]);
#pragma unroll
    for (int i = 0; i < 2; ++i)
#pragma unroll
      for (int k2 = 0; k2 < 2; ++k2)
#pragma unroll
        for (int j = 0; j < 4; ++j)
          acc[i][j] = __builtin_amdgcn_wmma_f32_16x16x32_bf16(
              false, af[i][k2], false, bfg[j][k2], (short)0, acc[i][j], false, false);
    __syncthreads();
  }

  // epilogue (C layout: VGPR r -> M = r + 8*(lane>=16), N = lane%16)
  const int nl = lane & 15;
  const int mh = (lane >> 4) * 8;
#pragma unroll
  for (int i = 0; i < 2; ++i)
#pragma unroll
    for (int j = 0; j < 4; ++j)
#pragma unroll
      for (int r = 0; r < 8; ++r) {
        int gm = bm + waveM + i * 16 + r + mh;
        int gn = bn + waveN + j * 16 + nl;
        if (gn < N) {
          size_t idx = (size_t)gm * N + gn;
          float v = acc[i][j][r];
          if (resid) v += resid[idx];
          if (outF) outF[idx] = v;
          if (outB) {
            float ov = v;
            if (gateMul) {                       // fused silu(gate) * up
              float g = gateMul[idx];
              ov = g / (1.0f + __expf(-g)) * v;
            }
            outB[idx] = f2bf(ov);
          }
        }
      }
}

// ---------------------------------------------------------------------------
// Flash-style causal attention with WMMA. 4 waves / block, each wave owns
// 16 query rows; block covers 64 rows of one (batch, head).
// Q/K layout: [B,NH,S,192] bf16.  V layout: [B,NH,S,128] bf16.
// K tile staged via TDM (contiguous 32x192 block as 6144x1 tile),
// double-buffered with issue-ahead.
// ---------------------------------------------------------------------------
__global__ __launch_bounds__(128)
void mla_attention(const u16* __restrict__ Qb, const u16* __restrict__ Kb,
                   const u16* __restrict__ Vb, u16* __restrict__ Out,
                   float scale)
{
  constexpr int S = 2048, HD = 192, VHD = 128, NH = 16;
  const int bh   = blockIdx.x;
  const int b    = bh >> 4;
  const int h    = bh & 15;
  const int tid  = threadIdx.x;
  const int wid  = tid >> 5;
  const int lane = tid & 31;
  const int mrow  = lane & 15;
  const int khalf = (lane >> 4) * 16;
  const int qbase = blockIdx.y * 64 + wid * 16;

  const u16* Qp = Qb + (size_t)bh * S * HD;
  const u16* Kp = Kb + (size_t)bh * S * HD;
  const u16* Vp = Vb + (size_t)bh * S * VHD;

  __shared__ u16 Ks[2][32 * HD];     // ping-pong K tiles [key][d]
  __shared__ u16 Vs[VHD * 32];       // transposed: [v][key]
  __shared__ u16 Ps[4][16 * 32];     // per-wave P scratch [m][key]

  // preload Q fragments (6 K-chunks of 32 over HD=192)
  v16bf qf[6];
#pragma unroll
  for (int c = 0; c < 6; ++c)
    qf[c] = *reinterpret_cast<const v16bf*>(
        Qp + (size_t)(qbase + mrow) * HD + c * 32 + khalf);

  v8f oacc[8];
#pragma unroll
  for (int t = 0; t < 8; ++t)
#pragma unroll
    for (int r = 0; r < 8; ++r) oacc[t][r] = 0.0f;
  float rowmax[8], rowsum[8];
#pragma unroll
  for (int r = 0; r < 8; ++r) { rowmax[r] = -1e30f; rowsum[r] = 0.0f; }

  const int klim = blockIdx.y * 64 + 64;
#if CDNA5_HAS_TDM
  if (wid == 0)   // pre-issue first K tile
    tdm_load_2d((unsigned)(size_t)(void*)Ks[0], Kp,
                32 * HD, 1, (unsigned)(S * HD), 1u, (unsigned)(32 * HD));
#endif
  for (int k0 = 0; k0 < klim; k0 += 32) {
    const int buf = (k0 >> 5) & 1;
    // ---- stage K tile (32 x 192 bf16, contiguous) ----
#if CDNA5_HAS_TDM
    if (wid == 0) {
      if (k0 + 32 < klim) {
        tdm_load_2d((unsigned)(size_t)(void*)Ks[buf ^ 1],
                    Kp + (size_t)(k0 + 32) * HD,
                    32 * HD, 1, (unsigned)(S * HD), 1u, (unsigned)(32 * HD));
        __builtin_amdgcn_s_wait_tensorcnt(1);
      } else {
        __builtin_amdgcn_s_wait_tensorcnt(0);
      }
    }
#else
    for (int c = tid; c < 32 * HD / 8; c += 128) {
      int key = c / 24, col = (c % 24) * 8;
      *reinterpret_cast<uint4*>(&Ks[buf][key * HD + col]) =
          *reinterpret_cast<const uint4*>(Kp + (size_t)(k0 + key) * HD + col);
    }
#endif
    // ---- stage V tile transposed (128 x 32) ----
    for (int c = tid; c < 32 * VHD / 8; c += 128) {
      int key = c / 16, col = (c % 16) * 8;
      u16 tmp[8];
      *reinterpret_cast<uint4*>(tmp) =
          *reinterpret_cast<const uint4*>(Vp + (size_t)(k0 + key) * VHD + col);
#pragma unroll
      for (int j = 0; j < 8; ++j) Vs[(col + j) * 32 + key] = tmp[j];
    }
    __syncthreads();

    // scores: two 16x16 tiles, each accumulated over 6 K-chunks
    v8f sc[2];
#pragma unroll
    for (int s = 0; s < 2; ++s) {
#pragma unroll
      for (int r = 0; r < 8; ++r) sc[s][r] = 0.0f;
#pragma unroll
      for (int c = 0; c < 6; ++c) {
        v16bf kf = *reinterpret_cast<const v16bf*>(
            &Ks[buf][(s * 16 + mrow) * HD + c * 32 + khalf]);
        sc[s] = __builtin_amdgcn_wmma_f32_16x16x32_bf16(
            false, qf[c], false, kf, (short)0, sc[s], false, false);
      }
    }

    // mask + scale, online softmax
    float p[2][8], tmax[8];
#pragma unroll
    for (int r = 0; r < 8; ++r) tmax[r] = -1e30f;
#pragma unroll
    for (int s = 0; s < 2; ++s)
#pragma unroll
      for (int r = 0; r < 8; ++r) {
        int qrow = qbase + r + ((lane >> 4) * 8);
        int kcol = k0 + s * 16 + mrow;
        float v = sc[s][r] * scale;
        if (kcol > qrow) v = -1e30f;
        p[s][r] = v;
        tmax[r] = fmaxf(tmax[r], v);
      }
#pragma unroll
    for (int r = 0; r < 8; ++r) {
      float m = tmax[r];
#pragma unroll
      for (int off = 8; off >= 1; off >>= 1)          // stays inside 16-lane half
        m = fmaxf(m, __shfl_xor(m, off, 32));
      float mnew  = fmaxf(rowmax[r], m);
      float alpha = __expf(rowmax[r] - mnew);
      float ts = 0.0f;
#pragma unroll
      for (int s = 0; s < 2; ++s) {
        float e = __expf(p[s][r] - mnew);
        p[s][r] = e;
        ts += e;
      }
#pragma unroll
      for (int off = 8; off >= 1; off >>= 1)
        ts += __shfl_xor(ts, off, 32);
      rowsum[r] = rowsum[r] * alpha + ts;
      rowmax[r] = mnew;
#pragma unroll
      for (int t = 0; t < 8; ++t) oacc[t][r] *= alpha;
    }

    // P -> per-wave LDS -> A-fragment layout (bf16)
#pragma unroll
    for (int s = 0; s < 2; ++s)
#pragma unroll
      for (int r = 0; r < 8; ++r) {
        int m = r + ((lane >> 4) * 8);
        Ps[wid][m * 32 + s * 16 + mrow] = f2bf(p[s][r]);
      }
    v16bf pf = *reinterpret_cast<const v16bf*>(&Ps[wid][mrow * 32 + khalf]);
#pragma unroll
    for (int t = 0; t < 8; ++t) {
      v16bf vf = *reinterpret_cast<const v16bf*>(&Vs[(t * 16 + mrow) * 32 + khalf]);
      oacc[t] = __builtin_amdgcn_wmma_f32_16x16x32_bf16(
          false, pf, false, vf, (short)0, oacc[t], false, false);
    }
    __syncthreads();
  }

  // write normalized output: attn[b, q, h*128 + v] bf16
#pragma unroll
  for (int t = 0; t < 8; ++t)
#pragma unroll
    for (int r = 0; r < 8; ++r) {
      int m = r + ((lane >> 4) * 8);
      int q = qbase + m;
      float v = oacc[t][r] / fmaxf(rowsum[r], 1e-20f);
      Out[((size_t)b * S + q) * (NH * VHD) + h * VHD + t * 16 + mrow] = f2bf(v);
    }
}

// ---------------------------------------------------------------------------
// Elementwise / norm kernels
// ---------------------------------------------------------------------------
__global__ __launch_bounds__(256)
void rmsnorm_bf16_kernel(const float* __restrict__ x, const float* __restrict__ w,
                         u16* __restrict__ out, int cols, int inStride, int outStride)
{
  const int row = blockIdx.x;
  const float* xr = x + (size_t)row * inStride;
  float ss = 0.0f;
  for (int c = threadIdx.x; c < cols; c += 256) { float v = xr[c]; ss += v * v; }
#pragma unroll
  for (int off = 16; off >= 1; off >>= 1) ss += __shfl_xor(ss, off, 32);
  __shared__ float red[8];
  __shared__ float sinv;
  if ((threadIdx.x & 31) == 0) red[threadIdx.x >> 5] = ss;
  __syncthreads();
  if (threadIdx.x == 0) {
    float tot = 0.0f;
    for (int i = 0; i < 8; ++i) tot += red[i];
    sinv = rsqrtf(tot / (float)cols + 1e-6f);
  }
  __syncthreads();
  float inv = sinv;
  u16* orow = out + (size_t)row * outStride;
  for (int c = threadIdx.x; c < cols; c += 256) orow[c] = f2bf(xr[c] * inv * w[c]);
}

__global__ void f32_to_bf16_kernel(const float* __restrict__ in, u16* __restrict__ out,
                                   long long n) {
  long long i = (long long)blockIdx.x * 256 + threadIdx.x;
  if (i < n) out[i] = f2bf(in[i]);
}

__device__ __forceinline__ float rope_val(const float* row, int dr, int s) {
  int j = dr & 31;
  float inv = __powf(10000.0f, -(float)(2 * j) / 64.0f);
  float cs, sn;
  __sincosf((float)s * inv, &cs, &sn);
  float x1 = row[j];
  float x2 = row[32 + j];
  return (dr < 32) ? (x1 * cs - x2 * sn) : (x2 * cs + x1 * sn);
}

// q f32 [B,S,NH,192] -> qb bf16 [B,NH,S,192] with RoPE on last 64 dims
__global__ void rope_pack_q_kernel(const float* __restrict__ q, u16* __restrict__ qb,
                                   long long total) {
  long long idx = (long long)blockIdx.x * 256 + threadIdx.x;
  if (idx >= total) return;
  constexpr int HD = 192, NH = 16, S = 2048;
  int d = (int)(idx % HD);
  long long t = idx / HD;
  int h = (int)(t % NH); t /= NH;
  int s = (int)(t % S);
  int b = (int)(t / S);
  const float* row = q + (((long long)b * S + s) * NH + h) * HD;
  float val = (d < 128) ? row[d] : rope_val(row + 128, d - 128, s);
  qb[(((long long)b * NH + h) * S + s) * HD + d] = f2bf(val);
}

// kv_exp f32 [B*S,4096] (k_nope cols h*256..+128) + kv_full rope cols 512..576
// -> kb bf16 [B,NH,S,192]
__global__ void pack_k_kernel(const float* __restrict__ kvexp,
                              const float* __restrict__ kvfull,
                              u16* __restrict__ kb, long long total) {
  long long idx = (long long)blockIdx.x * 256 + threadIdx.x;
  if (idx >= total) return;
  constexpr int HD = 192, NH = 16, S = 2048;
  int d = (int)(idx % HD);
  long long t = idx / HD;
  int h = (int)(t % NH); t /= NH;
  int s = (int)(t % S);
  int b = (int)(t / S);
  long long bs = (long long)b * S + s;
  float val;
  if (d < 128) val = kvexp[bs * 4096 + h * 256 + d];
  else         val = rope_val(kvfull + bs * 576 + 512, d - 128, s);
  kb[(((long long)b * NH + h) * S + s) * HD + d] = f2bf(val);
}

// v from kv_exp cols h*256+128..+256 -> vb bf16 [B,NH,S,128]
__global__ void pack_v_kernel(const float* __restrict__ kvexp, u16* __restrict__ vb,
                              long long total) {
  long long idx = (long long)blockIdx.x * 256 + threadIdx.x;
  if (idx >= total) return;
  constexpr int VHD = 128, NH = 16, S = 2048;
  int d = (int)(idx % VHD);
  long long t = idx / VHD;
  int h = (int)(t % NH); t /= NH;
  int s = (int)(t % S);
  int b = (int)(t / S);
  long long bs = (long long)b * S + s;
  vb[(((long long)b * NH + h) * S + s) * VHD + d] =
      f2bf(kvexp[bs * 4096 + h * 256 + 128 + d]);
}

// ---------------------------------------------------------------------------
// Driver
// ---------------------------------------------------------------------------
static void launch_gemm(const u16* A, const u16* B, int M, int N, int K,
                        const float* resid, float* outF, u16* outB,
                        const float* gateMul, hipStream_t s) {
  dim3 g((N + 127) / 128, M / 128), b(256);
  gemm_bf16_wmma<<<g, b, 0, s>>>(A, B, M, N, K, resid, outF, outB, gateMul);
}

extern "C" void kernel_launch(void* const* d_in, const int* in_sizes, int n_in,
                              void* d_out, int out_size, void* d_ws, size_t ws_size,
                              hipStream_t stream) {
  (void)in_sizes; (void)n_in; (void)out_size; (void)ws_size;
  const float* x        = (const float*)d_in[0];
  const float* attn_nw  = (const float*)d_in[1];
  const float* wq_down  = (const float*)d_in[2];
  const float* wq_up    = (const float*)d_in[3];
  const float* wkv_down = (const float*)d_in[4];
  const float* kv_nw    = (const float*)d_in[5];
  const float* wkv_up   = (const float*)d_in[6];
  const float* wout     = (const float*)d_in[7];
  const float* ffn_nw   = (const float*)d_in[8];
  const float* w_gate   = (const float*)d_in[9];
  const float* w_up     = (const float*)d_in[10];
  const float* w_down   = (const float*)d_in[11];

  constexpr long long BS = 4096;
  constexpr int HID = 2048, QR = 1536, NHHD = 3072, KVF = 576, KVR = 512,
                UPN = 4096, FFN = 5632, NHVD = 2048;

  char* wp = (char*)d_ws;
  auto alloc = [&](size_t bytes) -> void* {
    void* r = (void*)wp;
    wp += (bytes + 255) & ~(size_t)255;
    return r;
  };

  u16* wq_down_b  = (u16*)alloc((size_t)HID * QR * 2);
  u16* wq_up_b    = (u16*)alloc((size_t)QR * NHHD * 2);
  u16* wkv_down_b = (u16*)alloc((size_t)HID * KVF * 2);
  u16* wkv_up_b   = (u16*)alloc((size_t)KVR * UPN * 2);
  u16* wout_b     = (u16*)alloc((size_t)NHVD * HID * 2);
  u16* wgate_b    = (u16*)alloc((size_t)HID * FFN * 2);
  u16* wup_b      = (u16*)alloc((size_t)HID * FFN * 2);
  u16* wdown_b    = (u16*)alloc((size_t)FFN * HID * 2);

  u16*   hb     = (u16*)  alloc((size_t)BS * HID * 2);
  u16*   qdownb = (u16*)  alloc((size_t)BS * QR * 2);
  float* qf     = (float*)alloc((size_t)BS * NHHD * 4);
  float* kvfull = (float*)alloc((size_t)BS * KVF * 4);
  u16*   kvcb   = (u16*)  alloc((size_t)BS * KVR * 2);
  float* kvexp  = (float*)alloc((size_t)BS * UPN * 4);
  u16*   qb     = (u16*)  alloc((size_t)2 * 16 * 2048 * 192 * 2);
  u16*   kb     = (u16*)  alloc((size_t)2 * 16 * 2048 * 192 * 2);
  u16*   vb     = (u16*)  alloc((size_t)2 * 16 * 2048 * 128 * 2);
  u16*   attnb  = (u16*)  alloc((size_t)BS * NHVD * 2);
  float* out1   = (float*)alloc((size_t)BS * HID * 4);
  u16*   h2b    = (u16*)  alloc((size_t)BS * HID * 2);
  float* gatef  = (float*)alloc((size_t)BS * FFN * 4);
  u16*   h3b    = (u16*)  alloc((size_t)BS * FFN * 2);

  auto conv = [&](const float* src, u16* dst, long long n) {
    f32_to_bf16_kernel<<<dim3((unsigned)((n + 255) / 256)), dim3(256), 0, stream>>>(src, dst, n);
  };
  conv(wq_down,  wq_down_b,  (long long)HID * QR);
  conv(wq_up,    wq_up_b,    (long long)QR * NHHD);
  conv(wkv_down, wkv_down_b, (long long)HID * KVF);
  conv(wkv_up,   wkv_up_b,   (long long)KVR * UPN);
  conv(wout,     wout_b,     (long long)NHVD * HID);
  conv(w_gate,   wgate_b,    (long long)HID * FFN);
  conv(w_up,     wup_b,      (long long)HID * FFN);
  conv(w_down,   wdown_b,    (long long)FFN * HID);

  // attention path
  rmsnorm_bf16_kernel<<<dim3(4096), dim3(256), 0, stream>>>(x, attn_nw, hb, HID, HID, HID);
  launch_gemm(hb, wq_down_b, 4096, QR, HID, nullptr, nullptr, qdownb, nullptr, stream);
  launch_gemm(qdownb, wq_up_b, 4096, NHHD, QR, nullptr, qf, nullptr, nullptr, stream);
  launch_gemm(hb, wkv_down_b, 4096, KVF, HID, nullptr, kvfull, nullptr, nullptr, stream);
  rmsnorm_bf16_kernel<<<dim3(4096), dim3(256), 0, stream>>>(kvfull, kv_nw, kvcb, KVR, KVF, KVR);
  launch_gemm(kvcb, wkv_up_b, 4096, UPN, KVR, nullptr, kvexp, nullptr, nullptr, stream);

  const long long nq = (long long)2 * 2048 * 16 * 192;   // 12.6M
  const long long nv = (long long)2 * 2048 * 16 * 128;   // 8.4M
  rope_pack_q_kernel<<<dim3((unsigned)((nq + 255) / 256)), dim3(256), 0, stream>>>(qf, qb, nq);
  pack_k_kernel<<<dim3((unsigned)((nq + 255) / 256)), dim3(256), 0, stream>>>(kvexp, kvfull, kb, nq);
  pack_v_kernel<<<dim3((unsigned)((nv + 255) / 256)), dim3(256), 0, stream>>>(kvexp, vb, nv);

  mla_attention<<<dim3(32, 32), dim3(128), 0, stream>>>(qb, kb, vb, attnb,
                                                        1.0f / sqrtf(192.0f));

  launch_gemm(attnb, wout_b, 4096, HID, NHVD, x, out1, nullptr, nullptr, stream);

  // FFN path (silu fused into the up-projection GEMM epilogue)
  rmsnorm_bf16_kernel<<<dim3(4096), dim3(256), 0, stream>>>(out1, ffn_nw, h2b, HID, HID, HID);
  launch_gemm(h2b, wgate_b, 4096, FFN, HID, nullptr, gatef, nullptr, nullptr, stream);
  launch_gemm(h2b, wup_b,   4096, FFN, HID, nullptr, nullptr, h3b, gatef, stream);
  launch_gemm(h3b, wdown_b, 4096, HID, FFN, out1, (float*)d_out, nullptr, nullptr, stream);
}
